// CausalSelfAttention_78993038508726
// MI455X (gfx1250) — matrix operand order repaired and checked by reading, not compiled, not run on hardware
//
#include <hip/hip_runtime.h>

typedef _Float16 v16h __attribute__((ext_vector_type(16)));
typedef _Float16 v8h  __attribute__((ext_vector_type(8)));
typedef float    v8f  __attribute__((ext_vector_type(8)));
typedef unsigned int u32x4 __attribute__((ext_vector_type(4)));
typedef int          i32x8 __attribute__((ext_vector_type(8)));
typedef int          i32x4 __attribute__((ext_vector_type(4)));

#define B_  2
#define T_  2048
#define C_  1024
#define H_  16
#define DH_ 64

// ---------------------------------------------------------------------------
// WMMA fragment loaders (contiguous k order: k = 16*(lane/16) + s, wave32)
// ---------------------------------------------------------------------------

// A-fragment: 16x32 f16 tile from row-major memory, row stride lda (elements).
// lanes 0-15: row M=lane, K = 0..15; lanes 16-31: row M=lane-16, K = 16..31
__device__ __forceinline__ v16h load_a16x32(const _Float16* p, int lda) {
  const int lane = threadIdx.x & 31;
  const int row  = lane & 15;
  const int kb   = (lane >> 4) << 4;          // 0 or 16
  const _Float16* r = p + (size_t)row * lda + kb;
  v8h lo = *(const v8h*)(r);                  // K = kb+0..7
  v8h hi = *(const v8h*)(r + 8);              // K = kb+8..15
  v16h a;
#pragma unroll
  for (int i = 0; i < 8; ++i) { a[i] = lo[i]; a[8 + i] = hi[i]; }
  return a;
}

// B-fragment: 32x16 (KxN) tile from row-major KxN memory, row stride ldb.
__device__ __forceinline__ v16h load_b32x16(const _Float16* p, int ldb) {
  const int lane = threadIdx.x & 31;
  const int col  = lane & 15;
  const int kb   = (lane >> 4) << 4;          // 0 or 16
  v16h b;
#pragma unroll
  for (int h = 0; h < 16; ++h) b[h] = p[(size_t)(kb + h) * ldb + col];
  return b;
}

// B-fragment where memory holds B^T row-major (i.e. W[m][k]); 16 contiguous halves/lane.
__device__ __forceinline__ v16h load_bT32x16(const _Float16* p, int ldk) {
  const int lane = threadIdx.x & 31;
  const int col  = lane & 15;
  const int kb   = (lane >> 4) << 4;          // 0 or 16
  return *(const v16h*)(p + (size_t)col * ldk + kb);
}

// ---------------------------------------------------------------------------
// TDM: DMA a 2D f16 tile (tile_d1 rows x tile_d0 contiguous cols) into LDS.
// D# per CDNA5 ISA 8.3/8.4; amdgpu-toolchain 6-arg builtin; groups 2/3 zero (2D tile).
// ---------------------------------------------------------------------------
__device__ __forceinline__ void tdm_load_tile_f16(const _Float16* gptr, _Float16* ldsPtr,
                                                  unsigned tensor_d0, unsigned tensor_d1,
                                                  unsigned tile_d0, unsigned tile_d1,
                                                  unsigned stride0) {
  const unsigned lds_addr = (unsigned)(size_t)(void*)ldsPtr;  // low 32 bits = LDS offset
  const unsigned long long ga = (unsigned long long)(size_t)gptr;
  u32x4 g0;
  g0[0] = 1u;                                         // count=1, user descriptor
  g0[1] = lds_addr;                                   // lds_addr [63:32]
  g0[2] = (unsigned)(ga & 0xffffffffu);               // global_addr [95:64]
  g0[3] = (unsigned)((ga >> 32) & 0x1ffffffu) | (2u << 30);  // addr[56:32] | type=2
  i32x8 g1;
  g1[0] = (int)(1u << 16);                            // data_size=1 (2 bytes); mask=0
  g1[1] = (int)(tensor_d0 << 16);                     // tensor_dim0[15:0] -> [63:48]
  g1[2] = (int)((tensor_d0 >> 16) | (tensor_d1 << 16));
  g1[3] = (int)((tensor_d1 >> 16) | (tile_d0 << 16)); // tile_dim0 -> [127:112]
  g1[4] = (int)(tile_d1);                             // tile_dim1 [143:128]; tile_dim2=0
  g1[5] = (int)stride0;                               // tensor_dim0_stride[31:0]
  g1[6] = 0;                                          // stride[47:32]=0, dim1_stride lo=0
  g1[7] = 0;
  const i32x4 z4 = {0, 0, 0, 0};
  const i32x8 z8 = {0, 0, 0, 0, 0, 0, 0, 0};
  __builtin_amdgcn_tensor_load_to_lds(g0, g1, z4, z4, z8, 0);
}

// ---------------------------------------------------------------------------
// fp32 -> f16 conversion
// ---------------------------------------------------------------------------
__global__ void __launch_bounds__(256) f32_to_f16_kernel(const float* __restrict__ in,
                                                         _Float16* __restrict__ out, int n) {
  for (int i = blockIdx.x * blockDim.x + threadIdx.x; i < n; i += gridDim.x * blockDim.x)
    out[i] = (_Float16)in[i];
}

// ---------------------------------------------------------------------------
// out[n,m] = sum_k A[n,k] * W[m,k] + bias[m]    (A: NxK f16, W: MxK f16)
// Block (8 waves) owns a 128(n) x 64(m) tile; the shared 64x32 W k-slice is
// double-buffered in LDS via the Tensor Data Mover; wave w computes rows n0+16w.
// Grid must be exactly (N/128)*(M/64) blocks (barriers every k-step).
// ---------------------------------------------------------------------------
__global__ void __launch_bounds__(256)
gemm_xWt_wmma(const _Float16* __restrict__ A, const _Float16* __restrict__ W,
              const float* __restrict__ bias, _Float16* __restrict__ out16,
              float* __restrict__ out32, int N, int M, int K) {
  __shared__ __align__(128) _Float16 wtile[2][64 * 32];   // 2 x 4 KB double buffer

  const int lane = threadIdx.x & 31;
  const int warp = threadIdx.x >> 5;
  const int bn   = blockIdx.x % (N >> 7);
  const int bm   = blockIdx.x / (N >> 7);
  const int n0   = (bn << 7) + (warp << 4);
  const int m0   = bm << 6;

  const int kSteps = K >> 5;
  if (warp == 0)      // wave 0 drives the DMA; TENSORcnt is per-wave
    tdm_load_tile_f16(W + (size_t)m0 * K, wtile[0], (unsigned)K, (unsigned)M, 32u, 64u,
                      (unsigned)K);

  v8f acc[4] = {};
  for (int ks = 0; ks < kSteps; ++ks) {
    if (warp == 0) {
      if (ks + 1 < kSteps) {
        tdm_load_tile_f16(W + (size_t)m0 * K + 32 * (ks + 1), wtile[(ks + 1) & 1],
                          (unsigned)K, (unsigned)M, 32u, 64u, (unsigned)K);
        __builtin_amdgcn_s_wait_tensorcnt(1);  // tile ks resident (in-order completion)
      } else {
        __builtin_amdgcn_s_wait_tensorcnt(0);
      }
    }
    __syncthreads();                           // publish tile ks to all waves

    __builtin_prefetch(A + (size_t)n0 * K + 32 * ks + 128, 0, 1);
    const v16h a = load_a16x32(A + (size_t)n0 * K + 32 * ks, K);
    const _Float16* wt = wtile[ks & 1];
#pragma unroll
    for (int t = 0; t < 4; ++t) {
      const v16h b = load_bT32x16(wt + t * (16 * 32), 32);   // ds_load_b128 x2
      acc[t] = __builtin_amdgcn_wmma_f32_16x16x32_f16(false, a, false, b,
                                                      (short)0, acc[t], false, false);
    }
    __syncthreads();                           // all reads done before buffer reuse
  }

  const int col = lane & 15;
  const int rb  = (lane >> 4) << 3;            // C/D layout: row = rb + i, col = lane&15
#pragma unroll
  for (int t = 0; t < 4; ++t) {
    const float bv = bias[m0 + 16 * t + col];
#pragma unroll
    for (int i = 0; i < 8; ++i) {
      const float v = acc[t][i] + bv;
      const size_t idx = (size_t)(n0 + rb + i) * M + m0 + 16 * t + col;
      if (out32) out32[idx] = v;
      else       out16[idx] = (_Float16)v;
    }
  }
}

// ---------------------------------------------------------------------------
// Flash attention with ALiBi + causal mask.
// One wave handles 16 query rows of one (b,h). 32 keys per iteration:
//   4 WMMA for S (16x32), online softmax in C-fragment layout,
//   P re-laid out via per-wave LDS tile, 4 WMMA for P@V (16x64 accum).
// ---------------------------------------------------------------------------
__global__ void __launch_bounds__(256)
attn_flash_wmma(const _Float16* __restrict__ Q, const _Float16* __restrict__ KV,
                _Float16* __restrict__ Y) {
  __shared__ __align__(32) _Float16 plds[8 * 512];   // 8 waves * 16x32 f16 tile
  const int lane = threadIdx.x & 31;
  const int warp = threadIdx.x >> 5;
  const int wid  = blockIdx.x * 8 + warp;            // 0..4095
  const int b    = wid >> 11;
  const int h    = (wid >> 7) & 15;
  const int q0   = (wid & 127) << 4;

  const size_t bT = (size_t)b * T_;
  const _Float16* Qp = Q  + (bT + q0) * C_ + h * DH_;
  const _Float16* Kp = KV + bT * (2 * C_) + h * DH_;
  const _Float16* Vp = KV + bT * (2 * C_) + C_ + h * DH_;

  const float slope = (float)(h + 1) / (float)H_;    // ALiBi slope
  const int col = lane & 15;
  const int rb  = (lane >> 4) << 3;

  const v16h aq0 = load_a16x32(Qp + 0,  C_);
  const v16h aq1 = load_a16x32(Qp + 32, C_);

  v8f acc[4] = {};
  float m[8], l[8];
#pragma unroll
  for (int i = 0; i < 8; ++i) { m[i] = -1e30f; l[i] = 0.0f; }

  _Float16* P = plds + warp * 512;

  for (int jj = 0; jj < q0 + 16; jj += 32) {
    __builtin_prefetch(Kp + (size_t)(jj + 32) * (2 * C_), 0, 1);
    // ---- S = Q @ K^T  (16 queries x 32 keys) ----
    v8f s0 = {}, s1 = {};
    {
      v16h bk;
      bk = load_bT32x16(Kp + (size_t)jj * (2 * C_) + 0, 2 * C_);
      s0 = __builtin_amdgcn_wmma_f32_16x16x32_f16(false, aq0, false, bk, (short)0, s0, false, false);
      bk = load_bT32x16(Kp + (size_t)jj * (2 * C_) + 32, 2 * C_);
      s0 = __builtin_amdgcn_wmma_f32_16x16x32_f16(false, aq1, false, bk, (short)0, s0, false, false);
      bk = load_bT32x16(Kp + (size_t)(jj + 16) * (2 * C_) + 0, 2 * C_);
      s1 = __builtin_amdgcn_wmma_f32_16x16x32_f16(false, aq0, false, bk, (short)0, s1, false, false);
      bk = load_bT32x16(Kp + (size_t)(jj + 16) * (2 * C_) + 32, 2 * C_);
      s1 = __builtin_amdgcn_wmma_f32_16x16x32_f16(false, aq1, false, bk, (short)0, s1, false, false);
    }
    // ---- scale + ALiBi + causal mask (element (rb+i, col)) ----
#pragma unroll
    for (int i = 0; i < 8; ++i) {
      const int qi = q0 + rb + i;
      const int k0 = jj + col;
      const int k1 = jj + 16 + col;
      const float v0 = s0[i] * 0.125f + slope * (float)(k0 - qi);
      const float v1 = s1[i] * 0.125f + slope * (float)(k1 - qi);
      s0[i] = (k0 <= qi) ? v0 : -1e30f;
      s1[i] = (k1 <= qi) ? v1 : -1e30f;
    }
    // ---- online softmax: row reductions across 16-lane half-groups (wave32) ----
    float sc[8];
#pragma unroll
    for (int i = 0; i < 8; ++i) {
      float rmax = fmaxf(s0[i], s1[i]);
#pragma unroll
      for (int off = 1; off < 16; off <<= 1)
        rmax = fmaxf(rmax, __shfl_xor(rmax, off, 32));
      const float mn = fmaxf(m[i], rmax);
      sc[i] = __expf(m[i] - mn);
      const float p0 = __expf(s0[i] - mn);
      const float p1 = __expf(s1[i] - mn);
      float rs = p0 + p1;
#pragma unroll
      for (int off = 1; off < 16; off <<= 1)
        rs += __shfl_xor(rs, off, 32);
      l[i] = l[i] * sc[i] + rs;
      m[i] = mn;
      s0[i] = p0;
      s1[i] = p1;
    }
#pragma unroll
    for (int t = 0; t < 4; ++t)
#pragma unroll
      for (int i = 0; i < 8; ++i) acc[t][i] *= sc[i];

    // ---- P: C-fragment layout -> LDS -> A-fragment layout (in-wave, in-order LDS) ----
#pragma unroll
    for (int i = 0; i < 8; ++i) {
      P[(rb + i) * 32 + col]      = (_Float16)s0[i];
      P[(rb + i) * 32 + 16 + col] = (_Float16)s1[i];
    }
    __asm__ volatile("" ::: "memory");
    const v16h ap = load_a16x32(P, 32);
    __asm__ volatile("" ::: "memory");

    // ---- O += P @ V  (16 x 64) ----
#pragma unroll
    for (int t = 0; t < 4; ++t) {
      v16h bv = load_b32x16(Vp + (size_t)jj * (2 * C_) + 16 * t, 2 * C_);
      acc[t] = __builtin_amdgcn_wmma_f32_16x16x32_f16(false, ap, false, bv, (short)0, acc[t], false, false);
    }
  }

  // ---- normalize, store y as f16 in [B,T,H*Dh] (== [B,T,C]) ----
#pragma unroll
  for (int i = 0; i < 8; ++i) l[i] = 1.0f / l[i];
  _Float16* Yp = Y + (bT + q0) * C_ + h * DH_;
#pragma unroll
  for (int t = 0; t < 4; ++t)
#pragma unroll
    for (int i = 0; i < 8; ++i)
      Yp[(size_t)(rb + i) * C_ + 16 * t + col] = (_Float16)(acc[t][i] * l[i]);
}

// ---------------------------------------------------------------------------
// launch
// ---------------------------------------------------------------------------
extern "C" void kernel_launch(void* const* d_in, const int* in_sizes, int n_in,
                              void* d_out, int out_size, void* d_ws, size_t ws_size,
                              hipStream_t stream) {
  const float* x   = (const float*)d_in[0];
  // d_in[1] = freqs_cis (unused under ALiBi)
  const float* Wq  = (const float*)d_in[2];
  const float* bq  = (const float*)d_in[3];
  const float* Wkv = (const float*)d_in[4];
  const float* bkv = (const float*)d_in[5];
  const float* Wo  = (const float*)d_in[6];
  const float* bo  = (const float*)d_in[7];
  float* out = (float*)d_out;

  char* ws = (char*)d_ws;
  _Float16* x16   = (_Float16*)(ws);             //  8 MB : x          [4096,1024]
  _Float16* Wq16  = (_Float16*)(ws + 8388608);   //  2 MB : Wq         [1024,1024]
  _Float16* Wkv16 = (_Float16*)(ws + 10485760);  //  4 MB : Wkv        [2048,1024]
  _Float16* Wo16  = (_Float16*)(ws + 14680064);  //  2 MB : Wo         [1024,1024]
  _Float16* Q16   = (_Float16*)(ws + 16777216);  //  8 MB : Q          [4096,1024]
  _Float16* KV16  = (_Float16*)(ws + 25165824);  // 16 MB : KV         [4096,2048]
  _Float16* Y16   = (_Float16*)(ws + 41943040);  //  8 MB : attn out   [4096,1024]

  const int NT = B_ * T_;  // 4096 token rows

  f32_to_f16_kernel<<<2048, 256, 0, stream>>>(x,   x16,   NT * C_);
  f32_to_f16_kernel<<<2048, 256, 0, stream>>>(Wq,  Wq16,  C_ * C_);
  f32_to_f16_kernel<<<2048, 256, 0, stream>>>(Wkv, Wkv16, 2 * C_ * C_);
  f32_to_f16_kernel<<<2048, 256, 0, stream>>>(Wo,  Wo16,  C_ * C_);

  // Q = x @ Wq^T + bq   : (4096/128)*(1024/64) = 512 blocks (exact; no early return)
  gemm_xWt_wmma<<<512, 256, 0, stream>>>(x16, Wq16, bq, Q16, nullptr, NT, C_, C_);
  // KV = x @ Wkv^T + bkv: 32*32 = 1024 blocks
  gemm_xWt_wmma<<<1024, 256, 0, stream>>>(x16, Wkv16, bkv, KV16, nullptr, NT, 2 * C_, C_);
  // flash attention     : 2*16*128 = 4096 waves -> 512 blocks
  attn_flash_wmma<<<512, 256, 0, stream>>>(Q16, KV16, Y16);
  // out = y @ Wo^T + bo (f32 output)
  gemm_xWt_wmma<<<512, 256, 0, stream>>>(Y16, Wo16, bo, nullptr, out, NT, C_, C_);
}
